// Decoder_39427799777559
// MI455X (gfx1250) — compile-verified, matrix operand order checked by reading, not run on hardware
//
#include <hip/hip_runtime.h>
#include <hip/hip_bf16.h>

#define B_ 64
#define T_ 64
#define H_ 512
#define L_ 2
#define V_ 32000

typedef __attribute__((ext_vector_type(16))) __bf16 v16bf;
typedef __attribute__((ext_vector_type(8)))  float  v8f;
typedef __attribute__((ext_vector_type(4)))  unsigned int v4u;
typedef __attribute__((ext_vector_type(8)))  int v8i;
typedef __attribute__((ext_vector_type(4)))  int v4i;

#if defined(__has_builtin)
#if __has_builtin(__builtin_amdgcn_tensor_load_to_lds)
#define USE_TDM 1
#endif
#endif

union TileU { v4u u[2]; v16bf b; };

// Load one 16x32 bf16 tile in the CDNA5 WMMA A/B per-lane layout.
// Row-major source, row = base row + (lane&15), K-group selected by lane>>4.
__device__ __forceinline__ v16bf load_tile(const __bf16* __restrict__ base,
                                           int row0, int ld, int k0, int lane) {
  const int r = lane & 15;
  const int half = lane >> 4;
  const __bf16* p = base + (size_t)(row0 + r) * ld + k0 + half * 8;
  TileU t;
  t.u[0] = *reinterpret_cast<const v4u*>(p);        // K = k0+half*8 .. +7
  t.u[1] = *reinterpret_cast<const v4u*>(p + 16);   // K = k0+16+half*8 .. +7
  return t.b;
}

__device__ __forceinline__ v16bf load_tile_lds(const __bf16* base, int elem_off) {
  TileU t;
  t.u[0] = *reinterpret_cast<const v4u*>(base + elem_off);
  t.u[1] = *reinterpret_cast<const v4u*>(base + elem_off + 16);
  return t.b;
}

__device__ __forceinline__ v8f wmma_bf16(v16bf a, v16bf b, v8f c) {
  return __builtin_amdgcn_wmma_f32_16x16x32_bf16(
      /*neg_a=*/false, a, /*neg_b=*/false, b,
      /*c_mod=*/(short)0, c, /*reuse_a=*/false, /*reuse_b=*/false);
}

__device__ __forceinline__ float sigmoidf_(float x) {
  return 1.0f / (1.0f + __expf(-x));
}

// ---------------------------------------------------------------- embedding
__global__ void embed_relu_kernel(const int* __restrict__ target,
                                  const int* __restrict__ bosp,
                                  const float* __restrict__ embedding,
                                  __bf16* __restrict__ embbf) {
  size_t i = (size_t)blockIdx.x * blockDim.x + threadIdx.x;  // over T*B*H
  if (i >= (size_t)T_ * B_ * H_) return;
  int h = (int)(i % H_);
  size_t tb = i / H_;
  int b = (int)(tb % B_);
  int t = (int)(tb / B_);
  int id = (t == 0) ? bosp[0] : target[b * T_ + (t - 1)];
  float v = embedding[(size_t)id * H_ + h];
  v = v > 0.0f ? v : 0.0f;
  embbf[i] = (__bf16)v;
}

// ---------------------------------------------------------------- f32->bf16
__global__ void cvt_bf16_kernel(const float* __restrict__ in,
                                __bf16* __restrict__ out, long n) {
  long i = (long)blockIdx.x * blockDim.x + threadIdx.x;
  long stride = (long)gridDim.x * blockDim.x;
  for (; i < n; i += stride) out[i] = (__bf16)in[i];
}

// ---------------------------------------------------------------- h init
__global__ void init_h_kernel(const float* __restrict__ hidden,
                              float* __restrict__ hf32,
                              __bf16* __restrict__ hbf) {
  int i = blockIdx.x * blockDim.x + threadIdx.x;  // L*B*H
  float v = hidden[i];
  hf32[i] = v;                 // parity-0 slot
  hbf[i] = (__bf16)v;
}

// ---------------------------------------------------------------- final h
__global__ void copy_h_kernel(const float* __restrict__ hf32,
                              float* __restrict__ out) {
  int i = blockIdx.x * blockDim.x + threadIdx.x;  // L*B*H, parity 0
  out[i] = hf32[i];
}

// ---------------------------------------------------------------- grid barrier
__device__ __forceinline__ void grid_barrier(unsigned* bar, unsigned target) {
  __syncthreads();
  __threadfence();
  if (threadIdx.x == 0) {
    atomicAdd(bar, 1u);
    while (atomicAdd(bar, 0u) < target) __builtin_amdgcn_s_sleep(1);
  }
  __syncthreads();
  __threadfence();
}

// ---------------------------------------------------------------- GRU (persistent)
// 16 blocks x 256 threads = 128 waves; wave w -> (m_tile = w>>5, n_tile = w&31)
// covering the full [B=64, H=512] hidden state in 16x16 tiles.
__global__ __launch_bounds__(256, 1) void gru_kernel(
    const __bf16* __restrict__ embbf,   // [T,B,H]
    const __bf16* __restrict__ wihbf,   // [L,3H,H]
    const __bf16* __restrict__ whhbf,   // [L,3H,H]
    const float* __restrict__ b_ih,     // [L,3H]
    const float* __restrict__ b_hh,     // [L,3H]
    float* __restrict__ hf32,           // [2,L,B,H] ping-pong
    __bf16* __restrict__ hbf,           // [2,L,B,H] ping-pong
    __bf16* __restrict__ outsbf,        // [T,B,H]
    unsigned* __restrict__ bar) {
  const int lane = threadIdx.x & 31;
  const int wave = (int)((blockIdx.x * blockDim.x + threadIdx.x) >> 5);  // 0..127
  const int m0 = (wave >> 5) * 16;  // batch-row tile
  const int n0 = (wave & 31) * 16;  // hidden-col tile
  const int half = lane >> 4;
  const int nl = lane & 15;
  const size_t LBH = (size_t)L_ * B_ * H_;

  unsigned nbar = 0;
  for (int t = 0; t < T_; ++t) {
    const int p = t & 1;
    for (int l = 0; l < L_; ++l) {
      const __bf16* xsrc = (l == 0)
          ? (embbf + (size_t)t * B_ * H_)
          : (hbf + (size_t)(p ^ 1) * LBH + 0 * (size_t)B_ * H_);
      const __bf16* hsrc = hbf + (size_t)p * LBH + (size_t)l * B_ * H_;
      const __bf16* wih = wihbf + (size_t)l * 3 * H_ * H_;
      const __bf16* whh = whhbf + (size_t)l * 3 * H_ * H_;

      v8f gir = {}, giz = {}, gin = {}, ghr = {}, ghz = {}, ghn = {};
#pragma unroll 4
      for (int k0 = 0; k0 < H_; k0 += 32) {
        v16bf ax = load_tile(xsrc, m0, H_, k0, lane);
        v16bf ah = load_tile(hsrc, m0, H_, k0, lane);
        v16bf br = load_tile(wih, n0, H_, k0, lane);
        v16bf bz = load_tile(wih + (size_t)H_ * H_, n0, H_, k0, lane);
        v16bf bn = load_tile(wih + (size_t)2 * H_ * H_, n0, H_, k0, lane);
        v16bf cr = load_tile(whh, n0, H_, k0, lane);
        v16bf cz = load_tile(whh + (size_t)H_ * H_, n0, H_, k0, lane);
        v16bf cn = load_tile(whh + (size_t)2 * H_ * H_, n0, H_, k0, lane);
        gir = wmma_bf16(ax, br, gir);
        giz = wmma_bf16(ax, bz, giz);
        gin = wmma_bf16(ax, bn, gin);
        ghr = wmma_bf16(ah, cr, ghr);
        ghz = wmma_bf16(ah, cz, ghz);
        ghn = wmma_bf16(ah, cn, ghn);
      }

      const int col = n0 + nl;
      const float bir = b_ih[l * 3 * H_ + 0 * H_ + col];
      const float biz = b_ih[l * 3 * H_ + 1 * H_ + col];
      const float bin = b_ih[l * 3 * H_ + 2 * H_ + col];
      const float bhr = b_hh[l * 3 * H_ + 0 * H_ + col];
      const float bhz = b_hh[l * 3 * H_ + 1 * H_ + col];
      const float bhn = b_hh[l * 3 * H_ + 2 * H_ + col];

      const float* hold = hf32 + (size_t)p * LBH + (size_t)l * B_ * H_;
      float* hnew = hf32 + (size_t)(p ^ 1) * LBH + (size_t)l * B_ * H_;
      __bf16* hbw = hbf + (size_t)(p ^ 1) * LBH + (size_t)l * B_ * H_;

#pragma unroll
      for (int e = 0; e < 8; ++e) {
        const int row = m0 + e + 8 * half;
        const size_t off = (size_t)row * H_ + col;
        float r = sigmoidf_(gir[e] + bir + ghr[e] + bhr);
        float z = sigmoidf_(giz[e] + biz + ghz[e] + bhz);
        float n = tanhf(gin[e] + bin + r * (ghn[e] + bhn));
        float ho = hold[off];
        float hv = (1.0f - z) * n + z * ho;
        hnew[off] = hv;
        __bf16 hb = (__bf16)hv;
        hbw[off] = hb;
        if (l == L_ - 1) outsbf[((size_t)t * B_ + row) * H_ + col] = hb;
      }

      nbar += 1;
      grid_barrier(bar, nbar * gridDim.x);
    }
  }
}

// ---------------------------------------------------------------- projection
// logits[b,t,v] = outs[t,b,:] . W_out[v,:] + b_out[v]; row g = b*T + t.
// Wave tile = 16 rows x 80 cols (5 accumulators). Block = 8 waves = 128 rows,
// all sharing one 80-column W_out strip staged in LDS by the Tensor Data Mover.
#define LDS_ROWS 80
#define KHALF    256          // K elements staged per half
#define LDS_LDW  264          // 128 dwords + 4 pad dwords per row, in bf16 units

__global__ __launch_bounds__(256, 2) void proj_kernel(
    const __bf16* __restrict__ outsbf,  // [T,B,H]
    const __bf16* __restrict__ woutbf,  // [V,H]
    const float* __restrict__ b_out,    // [V]
    float* __restrict__ logits) {       // [B,T,V]
  __shared__ __align__(16) __bf16 ldsW[LDS_ROWS * LDS_LDW];  // ~42 KB

  const int lane = threadIdx.x & 31;
  const int wv = threadIdx.x >> 5;                 // 0..7
  const int m0 = (blockIdx.x * 8 + wv) * 16;       // row tile base
  const int n0 = blockIdx.y * 80;                  // 5 x 16 col tiles
  const int half = lane >> 4;
  const int nl = lane & 15;

  // A-row addressing: g = m0 + (lane&15); t = g % T, b = g / T (T = 64)
  const int g = m0 + nl;
  const int tt = g & (T_ - 1);
  const int bb = g >> 6;
  const __bf16* arow = outsbf + ((size_t)tt * B_ + bb) * H_;

  v8f acc[5] = {};
  for (int h2 = 0; h2 < 2; ++h2) {
    if (h2) __syncthreads();   // protect LDS being re-staged

    // ---- stage W_out[n0 .. n0+79, h2*256 .. +255] into LDS (padded rows)
#if defined(USE_TDM) && defined(__gfx1250__)
    if (threadIdx.x < 32) {    // wave 0 issues the TDM op (per-wave issue)
      const unsigned lds_base =
          (unsigned)(size_t)(&ldsW[0]);  // low 32 bits of flat addr == LDS offset
      const unsigned long long ga =
          (unsigned long long)(size_t)(woutbf + (size_t)n0 * H_ + h2 * KHALF);
      v4u g0;
      g0[0] = 1u;                                   // count = 1, user descriptor
      g0[1] = lds_base;                             // lds_addr
      g0[2] = (unsigned)(ga & 0xffffffffu);         // global_addr[31:0]
      g0[3] = (unsigned)((ga >> 32) & 0x1ffffffu)   // global_addr[56:32]
              | (2u << 30);                         // type = 2 (image)
      v8i g1;
      g1[0] = (int)((2u << 16)                      // data_size = 4 bytes
                    | (1u << 20)                    // pad_enable
                    | (6u << 22)                    // pad_interval = 128 dwords
                    | (3u << 25));                  // pad_amount   = 4 dwords
      g1[1] = (int)(128u << 16);                    // tensor_dim0 = 128 dwords
      g1[2] = (int)(32000u << 16);                  // tensor_dim1 = V rows
      g1[3] = (int)(128u << 16);                    // tile_dim0 = 128 dwords
      g1[4] = LDS_ROWS;                             // tile_dim1 = 80 rows
      g1[5] = 256;                                  // tensor_dim0_stride (dwords)
      g1[6] = 0;
      g1[7] = 0;
      v4i gz4 = {0, 0, 0, 0};                       // 2-D tensor: groups 2/3 unused
      v8i gz8 = {0, 0, 0, 0, 0, 0, 0, 0};
      __builtin_amdgcn_tensor_load_to_lds(g0, g1, gz4, gz4, gz8, 0);
      __builtin_amdgcn_s_wait_tensorcnt(0);
    }
#else
    // fallback: cooperative staged copy, same padded LDS layout
    for (int idx = threadIdx.x; idx < LDS_ROWS * (KHALF / 8); idx += 256) {
      const int row = idx >> 5;       // 32 x 16B chunks per row
      const int c = idx & 31;
      *reinterpret_cast<v4u*>(ldsW + row * LDS_LDW + c * 8) =
          *reinterpret_cast<const v4u*>(woutbf + (size_t)(n0 + row) * H_ +
                                        h2 * KHALF + c * 8);
    }
#endif
    __syncthreads();

    // ---- 8 K-chunks of WMMA against the staged strip
#pragma unroll 2
    for (int kk = 0; kk < KHALF; kk += 32) {
      const int k0 = h2 * KHALF + kk;
      TileU ta;
      const __bf16* pa = arow + k0 + half * 8;
      ta.u[0] = *reinterpret_cast<const v4u*>(pa);
      ta.u[1] = *reinterpret_cast<const v4u*>(pa + 16);
      v16bf a = ta.b;
      // speculative prefetch of the next A chunk (global_prefetch_b8)
      __builtin_prefetch(arow + k0 + 32, 0, 3);
#pragma unroll
      for (int i = 0; i < 5; ++i) {
        const int off = (i * 16 + nl) * LDS_LDW + kk + half * 8;
        v16bf bmat = load_tile_lds(ldsW, off);
        acc[i] = wmma_bf16(a, bmat, acc[i]);
      }
    }
  }

#pragma unroll
  for (int i = 0; i < 5; ++i) {
    const int col = n0 + i * 16 + nl;
    const float bias = b_out[col];
#pragma unroll
    for (int e = 0; e < 8; ++e) {
      const int row = m0 + e + 8 * half;
      logits[(size_t)row * V_ + col] = acc[i][e] + bias;
    }
  }
}

// ---------------------------------------------------------------- launch
extern "C" void kernel_launch(void* const* d_in, const int* in_sizes, int n_in,
                              void* d_out, int out_size, void* d_ws, size_t ws_size,
                              hipStream_t stream) {
  (void)in_sizes; (void)n_in; (void)out_size; (void)ws_size;
  // setup_inputs order:
  // 0 encoder_outputs, 1 hidden, 2 target_tensor, 3 embedding,
  // 4 W_ih, 5 W_hh, 6 b_ih, 7 b_hh, 8 W_out, 9 b_out, 10 bos_idx
  const float* hidden    = (const float*)d_in[1];
  const int*   target    = (const int*)d_in[2];
  const float* embedding = (const float*)d_in[3];
  const float* W_ih      = (const float*)d_in[4];
  const float* W_hh      = (const float*)d_in[5];
  const float* b_ih      = (const float*)d_in[6];
  const float* b_hh      = (const float*)d_in[7];
  const float* W_out     = (const float*)d_in[8];
  const float* b_out     = (const float*)d_in[9];
  const int*   bosp      = (const int*)d_in[10];

  float* logits  = (float*)d_out;                              // B*T*V
  float* h_final = (float*)d_out + (size_t)B_ * T_ * V_;       // L*B*H

  // workspace carve-up (256B aligned)
  char* ws = (char*)d_ws;
  size_t off = 0;
  auto carve = [&](size_t bytes) {
    char* p = ws + off;
    off += (bytes + 255) & ~(size_t)255;
    return p;
  };
  __bf16*   embbf  = (__bf16*)carve((size_t)T_ * B_ * H_ * 2);
  __bf16*   wihbf  = (__bf16*)carve((size_t)L_ * 3 * H_ * H_ * 2);
  __bf16*   whhbf  = (__bf16*)carve((size_t)L_ * 3 * H_ * H_ * 2);
  __bf16*   woutbf = (__bf16*)carve((size_t)V_ * H_ * 2);
  float*    hf32   = (float*)carve((size_t)2 * L_ * B_ * H_ * 4);
  __bf16*   hbf    = (__bf16*)carve((size_t)2 * L_ * B_ * H_ * 2);
  __bf16*   outsbf = (__bf16*)carve((size_t)T_ * B_ * H_ * 2);
  unsigned* bar    = (unsigned*)carve(256);

  (void)hipMemsetAsync(bar, 0, 256, stream);

  // 1. embedding gather + relu + bf16 (time-major)
  {
    long n = (long)T_ * B_ * H_;
    embed_relu_kernel<<<(unsigned)((n + 255) / 256), 256, 0, stream>>>(
        target, bosp, embedding, embbf);
  }
  // 2. weight conversions
  {
    long n = (long)L_ * 3 * H_ * H_;
    cvt_bf16_kernel<<<(unsigned)((n + 255) / 256), 256, 0, stream>>>(W_ih, wihbf, n);
    cvt_bf16_kernel<<<(unsigned)((n + 255) / 256), 256, 0, stream>>>(W_hh, whhbf, n);
    long m = (long)V_ * H_;
    cvt_bf16_kernel<<<16384, 256, 0, stream>>>(W_out, woutbf, m);
  }
  // 3. initial hidden state into parity-0 ping-pong slots
  init_h_kernel<<<(L_ * B_ * H_) / 256, 256, 0, stream>>>(hidden, hf32, hbf);

  // 4. persistent GRU recurrence (16 blocks, grid barrier between layer-steps)
  gru_kernel<<<16, 256, 0, stream>>>(embbf, wihbf, whhbf, b_ih, b_hh,
                                     hf32, hbf, outsbf, bar);

  // 5. output projection: 256 row-tiles / 8 per block = 32; 2000 col-tiles / 5 = 400
  {
    dim3 grid(32, 400);
    proj_kernel<<<grid, 256, 0, stream>>>(outsbf, woutbf, b_out, logits);
  }

  // 6. final hidden state (t=63 wrote parity 0)
  copy_h_kernel<<<(L_ * B_ * H_) / 256, 256, 0, stream>>>(hf32, h_final);
}
